// ChainCRF_41970420417366
// MI455X (gfx1250) — matrix operand order repaired and checked by reading, not compiled, not run on hardware
//
#include <hip/hip_runtime.h>

// ChainCRF fused forward for MI455X (gfx1250, wave32, WMMA).
//
// energy[b,t,i,j] = (x[b,t,:].(Wt[i*K+j,:]+Ws[j,:]) + bt[i*K+j]+bs[j]) * mask[b,t]
// then CRF forward recurrence (logsumexp) + target path score.
//
// Kernel plan:
//   crf_prep_w : fold Ws into Wt, convert to f16, pre-swizzle into WMMA
//                B-fragment lane order (one contiguous 32B load per lane).
//   crf_prep_b : fold bs into bt (f32 bias per GEMM column).
//   crf_fused  : 1 workgroup per batch; per 16-timestep chunk do a
//                16 x 4225 x 256 GEMM with v_wmma_f32_16x16x32_f16 into LDS,
//                then run 16 recurrence steps out of LDS. No HBM energy tensor.

typedef __attribute__((ext_vector_type(16))) _Float16 v16h;
typedef __attribute__((ext_vector_type(8)))  float    v8f;

#define Bsz      32
#define Lsz      512
#define Dsz      256
#define Ksz      65              // num_labels + pad
#define N_COLS   (Ksz * Ksz)     // 4225 GEMM output columns
#define NTILE    265             // ceil(4225 / 16)
#define KSTEPS   8               // 256 / 32
#define CHUNK    16              // timesteps per chunk == WMMA M
#define EPITCH   (NTILE * 16)    // 4240 floats per LDS energy row
#define NTHREADS 512
#define NWAVES   (NTHREADS / 32)

// ---------------------------------------------------------------------------
// Prep: folded+swizzled f16 weights.
// B-fragment k order of this code (contiguous, as A) for v_wmma_f32_16x16x32_f16 (32x16 K-major operand):
//   lane 0-15  -> N = lane,    K = 2*v + h  (halves 0..15 -> K 0..15)
//   lane 16-31 -> N = lane-16, K = 16 + (2*v + h)
// Stored linearly as [tile][kstep][lane][16 halves] so each lane reads 32
// contiguous bytes per fragment.
// ---------------------------------------------------------------------------
__global__ void crf_prep_w(const float* __restrict__ Ws,
                           const float* __restrict__ Wt,
                           _Float16* __restrict__ bswz) {
  int o = blockIdx.x * blockDim.x + threadIdx.x;
  const int total = NTILE * KSTEPS * 32 * 16;
  if (o >= total) return;
  int hi   = o & 15;
  int lane = (o >> 4) & 31;
  int ks   = (o >> 9) & 7;
  int tile = o >> 12;
  int n    = tile * 16 + (lane & 15);                 // GEMM column (W' row)
  int kabs = ks * 32 + hi + ((lane >= 16) ? 16 : 0);  // reduction index
  float v = 0.0f;
  if (n < N_COLS) v = Wt[n * Dsz + kabs] + Ws[(n % Ksz) * Dsz + kabs];
  bswz[o] = (_Float16)v;
}

__global__ void crf_prep_b(const float* __restrict__ bs,
                           const float* __restrict__ bt,
                           float* __restrict__ bias) {
  int n = blockIdx.x * blockDim.x + threadIdx.x;
  if (n < N_COLS) bias[n] = bt[n] + bs[n % Ksz];
}

// ---------------------------------------------------------------------------
// Fused GEMM + CRF recurrence. One workgroup per batch element.
// ---------------------------------------------------------------------------
__launch_bounds__(NTHREADS, 1)
__global__ void crf_fused(const float* __restrict__ x,
                          const v16h*  __restrict__ bswz,
                          const float* __restrict__ bias,
                          const int*   __restrict__ target,
                          const float* __restrict__ mask,
                          float* __restrict__ out) {
  // A fragments, swizzled in this code's contiguous k order (as B; 16x32 f16):
  //   lane 0-15: M=lane, K = 0..15 -> halves 0..15
  //   lane 16-31: M=lane-16, K offsets +16.
  __shared__ _Float16 aswz[KSTEPS * 32 * 16];   //   8 KB
  __shared__ float    e_lds[CHUNK * EPITCH];    // 271 KB energy tile
  __shared__ float    part_lds[2][80];          // double-buffered alpha
  __shared__ float    mask_s[CHUNK];

  const int b    = blockIdx.x;
  const int tid  = threadIdx.x;
  const int lane = tid & 31;
  const int wave = tid >> 5;

  const float* xb = x + (size_t)b * Lsz * Dsz;

  float tgtacc = 0.0f;  // path score (thread 0)
  int   prev   = 0;     // previous label (thread 0)
  int   pbuf   = 0;     // part[] buffer parity (uniform across block)

  for (int chunk = 0; chunk < Lsz / CHUNK; ++chunk) {
    const int t0 = chunk * CHUNK;

    // ---- 1) x chunk -> f16 swizzled A fragments in LDS -------------------
    for (int idx = tid; idx < CHUNK * Dsz; idx += NTHREADS) {
      int m = idx >> 8;           // timestep within chunk (Dsz == 256)
      int k = idx & 255;
      float v = xb[(t0 + m) * Dsz + k];
      int ks     = k >> 5;
      int krel   = k & 31;
      int grp    = krel >> 3;     // 0..3
      int within = krel & 7;
      int alane  = m + ((krel >= 16) ? 16 : 0);
      int ahi    = krel & 15;
      aswz[(ks * 32 + alane) * 16 + ahi] = (_Float16)v;
    }
    if (tid < CHUNK) mask_s[tid] = mask[b * Lsz + t0 + tid];
    __syncthreads();

    // ---- 2) WMMA GEMM: [16 x 256] x [256 x 4225] -> LDS energy tile ------
    v16h afrag[KSTEPS];
#pragma unroll
    for (int ks = 0; ks < KSTEPS; ++ks)
      afrag[ks] = *(const v16h*)&aswz[(ks * 32 + lane) * 16];

    for (int tile = wave; tile < NTILE; tile += NWAVES) {
      v8f acc = {};
#pragma unroll
      for (int ks = 0; ks < KSTEPS; ++ks) {
        v16h bfrag = bswz[(tile * KSTEPS + ks) * 32 + lane];
        acc = __builtin_amdgcn_wmma_f32_16x16x32_f16(
            false, afrag[ks], false, bfrag, (short)0, acc, false, false);
      }
      // C layout: VGPR r, lanes 0-15 -> M=r, lanes 16-31 -> M=r+8; N=lane%16
      int n = tile * 16 + (lane & 15);
      if (n < N_COLS) {
        float bia   = bias[n];
        int   mbase = (lane >= 16) ? 8 : 0;
#pragma unroll
        for (int r = 0; r < 8; ++r) {
          int m = mbase + r;
          e_lds[m * EPITCH + n] = (acc[r] + bia) * mask_s[m];
        }
      }
    }
    __syncthreads();

    // ---- 3) CRF forward recurrence over this chunk -----------------------
    for (int tl = 0; tl < CHUNK; ++tl) {
      const int t = t0 + tl;
      const float* e = &e_lds[tl * EPITCH];  // e[i*65 + j]
      if (t == 0) {
        // part0 = energy[0, pad, :]; tgt0 = energy[0, pad, y0]
        if (tid < Ksz) part_lds[0][tid] = e[(Ksz - 1) * Ksz + tid];
        if (tid == 0) {
          int y0 = target[b * Lsz];
          tgtacc = e[(Ksz - 1) * Ksz + y0];
          prev   = y0;
        }
        pbuf = 0;
      } else {
        const float* po = part_lds[pbuf];
        float*       pn = part_lds[pbuf ^ 1];
        if (tid < 4 * Ksz) {               // 4 lanes cooperate per label j
          int j = tid >> 2, s = tid & 3;
          float mx = -1e30f;
          for (int i = s; i < Ksz; i += 4)
            mx = fmaxf(mx, e[i * Ksz + j] + po[i]);
          mx = fmaxf(mx, __shfl_xor(mx, 1, 32));
          mx = fmaxf(mx, __shfl_xor(mx, 2, 32));
          float sum = 0.0f;
          for (int i = s; i < Ksz; i += 4)
            sum += __expf(e[i * Ksz + j] + po[i] - mx);
          sum += __shfl_xor(sum, 1, 32);
          sum += __shfl_xor(sum, 2, 32);
          if (s == 0) {
            float pnew = mx + __logf(sum);
            float mt   = mask_s[tl];
            pn[j] = po[j] + (pnew - po[j]) * mt;  // masked update
          }
        }
        if (tid == 0) {
          int y = target[b * Lsz + t];
          tgtacc += e[prev * Ksz + y];
          prev = y;
        }
        pbuf ^= 1;
      }
      __syncthreads();
    }
  }

  // ---- final: out[b] = logsumexp(part) - tgt -----------------------------
  if (tid == 0) {
    const float* p = part_lds[pbuf];
    float mx = -1e30f;
    for (int i = 0; i < Ksz; ++i) mx = fmaxf(mx, p[i]);
    float sum = 0.0f;
    for (int i = 0; i < Ksz; ++i) sum += __expf(p[i] - mx);
    out[b] = mx + __logf(sum) - tgtacc;
  }
}

// ---------------------------------------------------------------------------
extern "C" void kernel_launch(void* const* d_in, const int* in_sizes, int n_in,
                              void* d_out, int out_size, void* d_ws, size_t ws_size,
                              hipStream_t stream) {
  const float* x    = (const float*)d_in[0];
  const float* Ws   = (const float*)d_in[1];
  const float* bs   = (const float*)d_in[2];
  const float* Wt   = (const float*)d_in[3];
  const float* bt   = (const float*)d_in[4];
  const int*   tgt  = (const int*)d_in[5];
  const float* mask = (const float*)d_in[6];
  float*       out  = (float*)d_out;

  // Workspace: swizzled f16 folded weights (2.17 MB) + f32 bias (16.9 KB).
  _Float16* bswz = (_Float16*)d_ws;
  float*    bias = (float*)((char*)d_ws + (size_t)NTILE * KSTEPS * 32 * 16 * sizeof(_Float16));

  const int wtotal = NTILE * KSTEPS * 32 * 16;
  crf_prep_w<<<(wtotal + 255) / 256, 256, 0, stream>>>(Ws, Wt, bswz);
  crf_prep_b<<<(N_COLS + 255) / 256, 256, 0, stream>>>(bs, bt, bias);
  crf_fused<<<Bsz, NTHREADS, 0, stream>>>(x, (const v16h*)bswz, bias, tgt, mask, out);
  (void)in_sizes; (void)n_in; (void)out_size; (void)ws_size;
}
